// DeepseekV3MLA_60894046323246
// MI455X (gfx1250) — matrix operand order repaired and checked by reading, not compiled, not run on hardware
//
#include <hip/hip_runtime.h>
#include <math.h>

// ---------------- problem constants ----------------
constexpr int Bn   = 2;
constexpr int Tn   = 1024;
constexpr int HIDn = 4096;
constexpr int Hn   = 32;
constexpr int D_NOPE = 128;
constexpr int D_V  = 128;
constexpr int D_QK = 192;             // 128 + 64
constexpr int Q_RANK  = 1536;
constexpr int KV_RANK = 512;
constexpr int ROWS = Bn * Tn;         // 2048
#define ATT_SCALE 0.07216878364870322f   // 192^-0.5
#define NEG_INF   (-1e30f)

// ---------------- types / helpers ----------------
typedef __attribute__((ext_vector_type(16))) __bf16 bf16x16;
typedef __attribute__((ext_vector_type(8)))  __bf16 bf16x8;
typedef __attribute__((ext_vector_type(4)))  __bf16 bf16x4;
typedef __attribute__((ext_vector_type(8)))  float  f32x8;
typedef int v4i_t __attribute__((vector_size(16)));   // matches builtin proto

__device__ __forceinline__ __bf16 f2bf(float f) {
  unsigned u = __builtin_bit_cast(unsigned, f);
  unsigned r = u + 0x7FFFu + ((u >> 16) & 1u);   // round-to-nearest-even
  unsigned short h = (unsigned short)(r >> 16);
  return __builtin_bit_cast(__bf16, h);
}

__device__ __forceinline__ bf16x16 concat8(bf16x8 lo, bf16x8 hi) {
  return __builtin_shufflevector(lo, hi, 0,1,2,3,4,5,6,7,8,9,10,11,12,13,14,15);
}

__device__ __forceinline__ f32x8 wmma_bf16(bf16x16 a, bf16x16 b, f32x8 c) {
  return __builtin_amdgcn_wmma_f32_16x16x32_bf16(
      false, a, false, b, (short)0, c, false, false);
}

// ---- async global->LDS copy (CDNA5 ASYNCcnt path), with sync fallback ----
#define AS_GLOBAL __attribute__((address_space(1)))
#define AS_LDS    __attribute__((address_space(3)))

#if __has_builtin(__builtin_amdgcn_global_load_async_to_lds_b128)
__device__ __forceinline__ void copy16_g2l(const __bf16* g, __bf16* l) {
  void* gv = (void*)g;   // drop const, still generic
  void* lv = (void*)l;
  __builtin_amdgcn_global_load_async_to_lds_b128(
      (AS_GLOBAL v4i_t*)gv, (AS_LDS v4i_t*)lv, 0, 0);
}
__device__ __forceinline__ void copy_wait() {
#if __has_builtin(__builtin_amdgcn_s_wait_asynccnt)
  __builtin_amdgcn_s_wait_asynccnt(0);
#else
  asm volatile("s_wait_asynccnt 0" ::: "memory");
#endif
}
#else
__device__ __forceinline__ void copy16_g2l(const __bf16* g, __bf16* l) {
  *(bf16x8*)l = *(const bf16x8*)g;
}
__device__ __forceinline__ void copy_wait() {}
#endif

// ======================================================================
// fp32 -> bf16 elementwise convert (one-time; removes cvt from GEMM loop)
// ======================================================================
__global__ __launch_bounds__(256)
void cvt_bf16_kernel(const float* __restrict__ in, __bf16* __restrict__ out, int n) {
  int i = (blockIdx.x * 256 + threadIdx.x) * 4;
  if (i + 3 < n) {
    float4 v = *(const float4*)(in + i);
    bf16x4 o = { f2bf(v.x), f2bf(v.y), f2bf(v.z), f2bf(v.w) };
    *(bf16x4*)(out + i) = o;
  }
}

// ======================================================================
// bf16-in / fp32-out GEMM via WMMA.
// Block: 128 threads (4 wave32). Tile 128(M) x 64(N), K-step 32.
// Waves arranged 2x2: wave tile 64(M) x 32(N) => 8 WMMA / wave / K-step.
// A staged row-major (async b128 copies); B staged transposed Bs[n][k] so
// B fragments are two contiguous ds_load_b128 per lane.
// Requires: M%128==0, N%64==0, K%32==0, lda%8==0 (all true here).
// ======================================================================
__global__ __launch_bounds__(128)
void gemm_bf16_kernel(const __bf16* __restrict__ A, const __bf16* __restrict__ Bm,
                      float* __restrict__ C,
                      int M, int N, int K, int lda, int ldb, int ldc) {
  __shared__ __align__(16) __bf16 As[128][40];  // [m][k] (+8 pad)
  __shared__ __align__(16) __bf16 Bs[64][40];   // [n][k] transposed (+8 pad)

  const int tid  = threadIdx.x;
  const int lane = tid & 31;
  const int wv   = tid >> 5;
  const int bm   = blockIdx.y * 128;
  const int bn   = blockIdx.x * 64;
  const int wr   = (wv >> 1) * 64;   // wave M offset
  const int wc   = (wv & 1)  * 32;   // wave N offset

  // hoisted per-thread staging pointers (advance by constants per K-step)
  const __bf16* aSrc = A + (size_t)(bm + tid) * lda;                 // row tid
  const int brow = tid & 31, bcb = (tid >> 5) * 16;                  // B role
  const __bf16* bSrc = Bm + (size_t)brow * ldb + bn + bcb;
  const size_t bStep = (size_t)32 * ldb;

  // fragment geometry (loop-invariant)
  const int arow = lane & 15;
  const int kh   = (lane >> 4) * 8;    // A: K half select
  const int ko   = (lane >> 4) * 8;    // B: K half select

  f32x8 acc[4][2] = {};

  for (int kt = 0; kt < K; kt += 32) {
    // ---- stage A tile 128x32: each thread copies its row (4 x b128) ----
    #pragma unroll
    for (int j = 0; j < 4; ++j) copy16_g2l(aSrc + j * 8, &As[tid][j * 8]);
    // ---- stage B tile 32x64 transposed: 16 cols of one row per thread ----
    {
      bf16x8 v0 = *(const bf16x8*)(bSrc);
      bf16x8 v1 = *(const bf16x8*)(bSrc + 8);
      #pragma unroll
      for (int j = 0; j < 8; ++j) {
        Bs[bcb + j][brow]     = v0[j];
        Bs[bcb + 8 + j][brow] = v1[j];
      }
    }
    copy_wait();
    __syncthreads();

    bf16x16 af[4], bfr[2];
    #pragma unroll
    for (int mi = 0; mi < 4; ++mi)
      af[mi] = concat8(*(const bf16x8*)&As[wr + mi * 16 + arow][kh],
                       *(const bf16x8*)&As[wr + mi * 16 + arow][kh + 16]);
    #pragma unroll
    for (int ni = 0; ni < 2; ++ni)
      bfr[ni] = concat8(*(const bf16x8*)&Bs[wc + ni * 16 + arow][ko],
                        *(const bf16x8*)&Bs[wc + ni * 16 + arow][ko + 16]);
    #pragma unroll
    for (int mi = 0; mi < 4; ++mi)
      #pragma unroll
      for (int ni = 0; ni < 2; ++ni)
        acc[mi][ni] = wmma_bf16(af[mi], bfr[ni], acc[mi][ni]);

    __syncthreads();
    aSrc += 32; bSrc += bStep;
  }

  const int cn = lane & 15, hi = lane >> 4;
  #pragma unroll
  for (int mi = 0; mi < 4; ++mi)
    #pragma unroll
    for (int ni = 0; ni < 2; ++ni)
      #pragma unroll
      for (int r = 0; r < 8; ++r) {
        int row = bm + wr + mi * 16 + r + 8 * hi;
        int col = bn + wc + ni * 16 + cn;
        C[(size_t)row * ldc + col] = acc[mi][ni][r];
      }
}

// ======================================================================
// RMSNorm: fp32 in (strided), bf16 out (dense). One block per row.
// ======================================================================
__global__ __launch_bounds__(256)
void rmsnorm_bf16_kernel(const float* __restrict__ x, const float* __restrict__ w,
                         __bf16* __restrict__ y, int cols, int xstride, int ystride) {
  const float* p = x + (size_t)blockIdx.x * xstride;
  __bf16* q = y + (size_t)blockIdx.x * ystride;
  float ss = 0.f;
  for (int c = threadIdx.x; c < cols; c += 256) { float v = p[c]; ss += v * v; }
  #pragma unroll
  for (int d = 16; d >= 1; d >>= 1) ss += __shfl_xor(ss, d);
  __shared__ float red[8];
  if ((threadIdx.x & 31) == 0) red[threadIdx.x >> 5] = ss;
  __syncthreads();
  float tot = 0.f;
  #pragma unroll
  for (int i = 0; i < 8; ++i) tot += red[i];
  float inv = rsqrtf(tot / (float)cols + 1e-6f);
  for (int c = threadIdx.x; c < cols; c += 256) q[c] = f2bf(p[c] * inv * w[c]);
}

// ======================================================================
// RoPE + pack to bf16 head-major tensors. One block per (b,t).
// ======================================================================
__global__ __launch_bounds__(256)
void rope_pack_kernel(const float* __restrict__ q,       // [ROWS, H*192] fp32
                      const float* __restrict__ kv_out,  // [ROWS, 576]   fp32
                      const float* __restrict__ kv_exp,  // [ROWS, H*256] fp32
                      const int*   __restrict__ positions,
                      __bf16* __restrict__ qp, __bf16* __restrict__ kp,
                      __bf16* __restrict__ vp) {
  const int bid = blockIdx.x;            // b*T + t
  const int b = bid / Tn, t = bid - b * Tn;
  const int tid = threadIdx.x;
  const float pos = (float)positions[bid];

  __shared__ float kpe[64];
  if (tid < 32) {
    float inv = __powf(10000.f, -(float)tid / 32.f);
    float ang = pos * inv, c = __cosf(ang), s = __sinf(ang);
    float x1 = kv_out[(size_t)bid * 576 + 512 + tid];
    float x2 = kv_out[(size_t)bid * 576 + 544 + tid];
    kpe[tid]      = x1 * c - x2 * s;
    kpe[tid + 32] = x1 * s + x2 * c;
  }
  __syncthreads();

  for (int i = tid; i < Hn * D_QK; i += 256) {
    int h = i / D_QK, d = i - h * D_QK;
    const float* qrow = q + (size_t)bid * (Hn * D_QK) + h * D_QK;
    float val;
    if (d < D_NOPE) {
      val = qrow[d];
    } else {
      int dd = d - D_NOPE, pi = dd & 31;
      float inv = __powf(10000.f, -(float)pi / 32.f);
      float ang = pos * inv, c = __cosf(ang), s = __sinf(ang);
      float x1 = qrow[128 + pi], x2 = qrow[160 + pi];
      val = (dd < 32) ? (x1 * c - x2 * s) : (x1 * s + x2 * c);
    }
    qp[((size_t)(b * Hn + h) * Tn + t) * D_QK + d] = f2bf(val);
  }
  for (int i = tid; i < Hn * D_QK; i += 256) {
    int h = i / D_QK, d = i - h * D_QK;
    float val = (d < D_NOPE) ? kv_exp[(size_t)bid * (Hn * 256) + h * 256 + d]
                             : kpe[d - D_NOPE];
    kp[((size_t)(b * Hn + h) * Tn + t) * D_QK + d] = f2bf(val);
  }
  for (int i = tid; i < Hn * D_V; i += 256) {
    int h = i / D_V, d = i - h * D_V;
    vp[((size_t)(b * Hn + h) * Tn + t) * D_V + d] =
        f2bf(kv_exp[(size_t)bid * (Hn * 256) + h * 256 + 128 + d]);
  }
}

// ======================================================================
// Flash-style causal attention. Grid: (B*H, T/64). Block: 128 (4 waves).
// Wave w owns query rows [qbase + 16w, +16). KV tiles of 32 staged in LDS
// (K via async global->LDS copies, V transposed through VGPRs).
// Output written directly as bf16 for the final WMMA GEMM.
// ======================================================================
__global__ __launch_bounds__(128)
void mla_attn_kernel(const __bf16* __restrict__ qp, const __bf16* __restrict__ kp,
                     const __bf16* __restrict__ vp, const int* __restrict__ amask,
                     __bf16* __restrict__ out /* [B,T,H*128] bf16 */) {
  const int bh = blockIdx.x;
  const int b  = bh >> 5, h = bh & 31;
  const int qbase = blockIdx.y * 64;
  const int tid = threadIdx.x, lane = tid & 31, wv = tid >> 5;
  const int qrow0 = qbase + wv * 16;
  const int ln = lane & 15, hi = lane >> 4;

  __shared__ __align__(16) __bf16 Ks[32][200];     // [kv][d]   (+8 pad)
  __shared__ __align__(16) __bf16 Vst[128][40];    // [d][kv]   transposed
  __shared__ __align__(16) __bf16 Ps[4][16][40];   // per-wave P scratch

  const __bf16* Qg = qp + (size_t)bh * Tn * D_QK;
  const __bf16* Kg = kp + (size_t)bh * Tn * D_QK;
  const __bf16* Vg = vp + (size_t)bh * Tn * D_V;
  const int* amrow = amask + b * Tn;

  // Q fragments: 6 chunks of K-dim 32, rows qrow0..+15
  bf16x16 qa[6];
  {
    const __bf16* qr = Qg + (size_t)(qrow0 + ln) * D_QK;
    const int khq = hi * 8;
    #pragma unroll
    for (int c = 0; c < 6; ++c)
      qa[c] = concat8(*(const bf16x8*)(qr + c * 32 + khq),
                      *(const bf16x8*)(qr + c * 32 + khq + 16));
  }

  f32x8 o[8] = {};
  float m[8], l[8];
  #pragma unroll
  for (int r = 0; r < 8; ++r) { m[r] = -3.0e38f; l[r] = 0.f; }

  const int nkt = (qbase + 64) / 32;   // uniform across waves (causal bound)
  for (int kt = 0; kt < nkt; ++kt) {
    const int kvbase = kt * 32;
    #pragma unroll
    for (int n = 0; n < 6; ++n) {
      int i = tid * 8 + n * 1024;               // 32*192 = 6144 elems
      int rr = i / 192, cc = i - rr * 192;
      copy16_g2l(Kg + (size_t)(kvbase + rr) * D_QK + cc, &Ks[rr][cc]);
    }
    #pragma unroll
    for (int n = 0; n < 4; ++n) {
      int i = tid * 8 + n * 1024;               // 32*128 = 4096 elems
      int rr = i >> 7, cc = i & 127;
      bf16x8 v = *(const bf16x8*)(Vg + (size_t)(kvbase + rr) * D_V + cc);
      #pragma unroll
      for (int j = 0; j < 8; ++j) Vst[cc + j][rr] = v[j];
    }
    copy_wait();
    __syncthreads();

    // ---- S = Q K^T (two 16-col subtiles) ----
    f32x8 s0 = {}, s1 = {};
    {
      const int ko = hi * 8;
      #pragma unroll
      for (int c = 0; c < 6; ++c) {
        bf16x16 b0 = concat8(*(const bf16x8*)&Ks[ln][c * 32 + ko],
                             *(const bf16x8*)&Ks[ln][c * 32 + ko + 16]);
        bf16x16 b1 = concat8(*(const bf16x8*)&Ks[ln + 16][c * 32 + ko],
                             *(const bf16x8*)&Ks[ln + 16][c * 32 + ko + 16]);
        s0 = wmma_bf16(qa[c], b0, s0);
        s1 = wmma_bf16(qa[c], b1, s1);
      }
    }

    // ---- scale + causal/padding mask ----
    const int kvc = kvbase + ln;
    const int am0 = amrow[kvc];
    const int am1 = amrow[kvc + 16];
    #pragma unroll
    for (int r = 0; r < 8; ++r) {
      const int qi = qrow0 + r + 8 * hi;
      float v0 = s0[r] * ATT_SCALE, v1 = s1[r] * ATT_SCALE;
      if (kvc > qi || am0 == 0)      v0 = NEG_INF;
      if (kvc + 16 > qi || am1 == 0) v1 = NEG_INF;
      s0[r] = v0; s1[r] = v1;
    }

    // ---- online softmax (16-lane row reductions; wave32) ----
    #pragma unroll
    for (int r = 0; r < 8; ++r) {
      float mx = fmaxf(s0[r], s1[r]);
      #pragma unroll
      for (int d = 1; d < 16; d <<= 1) mx = fmaxf(mx, __shfl_xor(mx, d));
      float mnew = fmaxf(m[r], mx);
      float corr = __expf(m[r] - mnew);
      float e0 = __expf(s0[r] - mnew);
      float e1 = __expf(s1[r] - mnew);
      float rs = e0 + e1;
      #pragma unroll
      for (int d = 1; d < 16; d <<= 1) rs += __shfl_xor(rs, d);
      l[r] = l[r] * corr + rs;
      m[r] = mnew;
      #pragma unroll
      for (int g = 0; g < 8; ++g) o[g][r] *= corr;
      Ps[wv][r + 8 * hi][ln]      = f2bf(e0);
      Ps[wv][r + 8 * hi][16 + ln] = f2bf(e1);
    }

    // ---- O += P (16x32) x V (32x128) via per-wave LDS re-layout ----
    {
      const int khp = hi * 8;
      bf16x16 pa = concat8(*(const bf16x8*)&Ps[wv][ln][khp],
                           *(const bf16x8*)&Ps[wv][ln][khp + 16]);
      const int ko = hi * 8;
      #pragma unroll
      for (int g = 0; g < 8; ++g) {
        bf16x16 vb = concat8(*(const bf16x8*)&Vst[g * 16 + ln][ko],
                             *(const bf16x8*)&Vst[g * 16 + ln][ko + 16]);
        o[g] = wmma_bf16(pa, vb, o[g]);
      }
    }
    __syncthreads();
  }

  #pragma unroll
  for (int r = 0; r < 8; ++r) {
    const int qi = qrow0 + r + 8 * hi;
    const float linv = (l[r] > 0.f) ? 1.f / l[r] : 0.f;
    __bf16* orow = out + ((size_t)(b * Tn + qi) * Hn + h) * D_V;
    #pragma unroll
    for (int g = 0; g < 8; ++g) orow[g * 16 + ln] = f2bf(o[g][r] * linv);
  }
}

// ======================================================================
// Host launcher
// ======================================================================
extern "C" void kernel_launch(void* const* d_in, const int* in_sizes, int n_in,
                              void* d_out, int out_size, void* d_ws, size_t ws_size,
                              hipStream_t stream) {
  (void)in_sizes; (void)n_in; (void)out_size; (void)ws_size;
  const float* x        = (const float*)d_in[0];
  const float* wq_a     = (const float*)d_in[1];
  const float* q_norm_w = (const float*)d_in[2];
  const float* wq_b     = (const float*)d_in[3];
  const float* wkv_a    = (const float*)d_in[4];
  const float* kv_norm_w= (const float*)d_in[5];
  const float* wkv_b    = (const float*)d_in[6];
  const float* wo       = (const float*)d_in[7];
  const int*   amask    = (const int*)d_in[8];
  const int*   positions= (const int*)d_in[9];
  float* out = (float*)d_out;

  // workspace (~322 MB, 256B aligned)
  char* ws = (char*)d_ws;
  size_t off = 0;
  auto carve = [&](size_t bytes) { char* p = ws + off; off += (bytes + 255) & ~(size_t)255; return p; };
  // fp32 intermediates
  float*  q_down  = (float*)carve((size_t)ROWS * Q_RANK * 4);
  float*  qbuf    = (float*)carve((size_t)ROWS * Hn * D_QK * 4);
  float*  kv_out  = (float*)carve((size_t)ROWS * 576 * 4);
  float*  kv_exp  = (float*)carve((size_t)ROWS * Hn * 256 * 4);
  // bf16 operands
  __bf16* x_bf    = (__bf16*)carve((size_t)ROWS * HIDn * 2);
  __bf16* wqa_bf  = (__bf16*)carve((size_t)HIDn * Q_RANK * 2);
  __bf16* wqb_bf  = (__bf16*)carve((size_t)Q_RANK * Hn * D_QK * 2);
  __bf16* wkva_bf = (__bf16*)carve((size_t)HIDn * 576 * 2);
  __bf16* wkvb_bf = (__bf16*)carve((size_t)KV_RANK * Hn * 256 * 2);
  __bf16* wo_bf   = (__bf16*)carve((size_t)HIDn * HIDn * 2);
  __bf16* qdn_bf  = (__bf16*)carve((size_t)ROWS * Q_RANK * 2);
  __bf16* kvn_bf  = (__bf16*)carve((size_t)ROWS * KV_RANK * 2);
  __bf16* q_pack  = (__bf16*)carve((size_t)Bn * Hn * Tn * D_QK * 2);
  __bf16* k_pack  = (__bf16*)carve((size_t)Bn * Hn * Tn * D_QK * 2);
  __bf16* v_pack  = (__bf16*)carve((size_t)Bn * Hn * Tn * D_V * 2);
  __bf16* attn_bf = (__bf16*)carve((size_t)ROWS * Hn * D_V * 2);

  auto cvt = [&](const float* src, __bf16* dst, size_t n) {
    cvt_bf16_kernel<<<(unsigned)(n / 1024), 256, 0, stream>>>(src, dst, (int)n);
  };
  // 0) one-time bf16 conversion of activations + weights
  cvt(x,     x_bf,    (size_t)ROWS * HIDn);
  cvt(wq_a,  wqa_bf,  (size_t)HIDn * Q_RANK);
  cvt(wq_b,  wqb_bf,  (size_t)Q_RANK * Hn * D_QK);
  cvt(wkv_a, wkva_bf, (size_t)HIDn * 576);
  cvt(wkv_b, wkvb_bf, (size_t)KV_RANK * Hn * 256);
  cvt(wo,    wo_bf,   (size_t)HIDn * HIDn);

  // 1) q_down = x @ wq_a
  gemm_bf16_kernel<<<dim3(Q_RANK / 64, ROWS / 128), 128, 0, stream>>>(
      x_bf, wqa_bf, q_down, ROWS, Q_RANK, HIDn, HIDn, Q_RANK, Q_RANK);
  // 2) RMSNorm(q_down) -> bf16
  rmsnorm_bf16_kernel<<<ROWS, 256, 0, stream>>>(q_down, q_norm_w, qdn_bf,
                                                Q_RANK, Q_RANK, Q_RANK);
  // 3) q = qdn @ wq_b
  gemm_bf16_kernel<<<dim3(Hn * D_QK / 64, ROWS / 128), 128, 0, stream>>>(
      qdn_bf, wqb_bf, qbuf, ROWS, Hn * D_QK, Q_RANK, Q_RANK, Hn * D_QK, Hn * D_QK);
  // 4) kv_out = x @ wkv_a
  gemm_bf16_kernel<<<dim3(576 / 64, ROWS / 128), 128, 0, stream>>>(
      x_bf, wkva_bf, kv_out, ROWS, 576, HIDn, HIDn, 576, 576);
  // 5) RMSNorm on first 512 cols (stride 576) -> dense bf16
  rmsnorm_bf16_kernel<<<ROWS, 256, 0, stream>>>(kv_out, kv_norm_w, kvn_bf,
                                                KV_RANK, 576, KV_RANK);
  // 6) kv_exp = kvn @ wkv_b
  gemm_bf16_kernel<<<dim3(Hn * 256 / 64, ROWS / 128), 128, 0, stream>>>(
      kvn_bf, wkvb_bf, kv_exp, ROWS, Hn * 256, KV_RANK, KV_RANK, Hn * 256, Hn * 256);
  // 7) RoPE + pack Q/K/V head-major bf16
  rope_pack_kernel<<<ROWS, 256, 0, stream>>>(
      qbuf, kv_out, kv_exp, positions, q_pack, k_pack, v_pack);
  // 8) causal attention -> bf16
  mla_attn_kernel<<<dim3(Bn * Hn, Tn / 64), 128, 0, stream>>>(
      q_pack, k_pack, v_pack, amask, attn_bf);
  // 9) out = attn @ wo
  gemm_bf16_kernel<<<dim3(HIDn / 64, ROWS / 128), 128, 0, stream>>>(
      attn_bf, wo_bf, out, ROWS, HIDn, Hn * D_V, Hn * D_V, HIDn, HIDn);
}
